// GATEAUNet_81698867904520
// MI455X (gfx1250) — matrix operand order repaired and checked by reading, not compiled, not run on hardware
//
#include <hip/hip_runtime.h>
#include <hip/hip_bf16.h>

// ---------------------------------------------------------------------------
// Types for WMMA
// ---------------------------------------------------------------------------
typedef __bf16 v16bf __attribute__((ext_vector_type(16)));
typedef float  v8f   __attribute__((ext_vector_type(8)));
typedef unsigned int u32x4 __attribute__((ext_vector_type(4)));

union Frag16 {            // 32 bytes: 16 bf16 == 2 x 16B vector loads
    u32x4 u[2];
    v16bf v;
};

#define D_MODEL 128
#define N_CELLS 121
#define BATCH   256
#define TOKENS  (BATCH * N_CELLS)      // 30976, multiple of 32
#define FF_DIM  512
#define N_LAYERS 6

// ---------------------------------------------------------------------------
// Helpers
// ---------------------------------------------------------------------------
__device__ __forceinline__ unsigned short f2bf(float x) {
    unsigned int u = __float_as_uint(x);
    unsigned int r = u + 0x7fffu + ((u >> 16) & 1u);   // RNE
    return (unsigned short)(r >> 16);
}

__device__ __forceinline__ float gelu_f(float x) {
    return 0.5f * x * (1.0f + erff(x * 0.70710678118654752f));
}

__device__ __forceinline__ float wave_sum(float v) {
    #pragma unroll
    for (int o = 16; o > 0; o >>= 1) v += __shfl_xor(v, o, 32);
    return v;
}

// ---------------------------------------------------------------------------
// Weight pack: f32 row-major (L,K,N) -> bf16 WMMA B-fragment layout.
// Dest element id = ((((layer*Nt + nt)*Kt + kt)*32 + lane)*16 + j)
// maps to source (k = kt*32 + (j>>3)*16 + (lane>>4)*8 + (j&7), n = nt*16 + (lane&15)).
// ---------------------------------------------------------------------------
__global__ void pack_w_kernel(const float* __restrict__ W,
                              unsigned short* __restrict__ Wpk,
                              int L, int K, int N) {
    size_t id = (size_t)blockIdx.x * 256 + threadIdx.x;
    size_t total = (size_t)L * K * N;
    if (id >= total) return;
    int j    = (int)(id & 15);
    int lane = (int)((id >> 4) & 31);
    size_t rest = id >> 9;
    int Kt = K >> 5, Nt = N >> 4;
    int kt = (int)(rest % Kt); rest /= Kt;
    int nt = (int)(rest % Nt);
    int layer = (int)(rest / Nt);
    int k = kt * 32 + (j >> 3) * 16 + (lane >> 4) * 8 + (j & 7);
    int n = nt * 16 + (lane & 15);
    Wpk[id] = f2bf(W[((size_t)layer * K + k) * N + n]);
}

// ---------------------------------------------------------------------------
// Encoder: h = gelu(LN(x @ enc_W + enc_b)) ; one 128-thread block per token.
// ---------------------------------------------------------------------------
__global__ void encoder_kernel(const float* __restrict__ x,
                               const float* __restrict__ W,
                               const float* __restrict__ b,
                               const float* __restrict__ g,
                               const float* __restrict__ beta,
                               float* __restrict__ h) {
    int t = blockIdx.x;
    int d = threadIdx.x;                       // 0..127
    const float* xr = x + (size_t)t * 6;
    float acc = b[d];
    #pragma unroll
    for (int k = 0; k < 6; ++k) acc += xr[k] * W[k * D_MODEL + d];

    __shared__ float ws[4], ws2[4];
    int wave = d >> 5, lane = d & 31;
    float s = wave_sum(acc);
    if (lane == 0) ws[wave] = s;
    __syncthreads();
    float mean = (ws[0] + ws[1] + ws[2] + ws[3]) * (1.0f / 128.0f);
    float diff = acc - mean;
    float s2 = wave_sum(diff * diff);
    if (lane == 0) ws2[wave] = s2;
    __syncthreads();
    float var = (ws2[0] + ws2[1] + ws2[2] + ws2[3]) * (1.0f / 128.0f);
    float y = diff * rsqrtf(var + 1e-5f) * g[d] + beta[d];
    h[(size_t)t * D_MODEL + d] = gelu_f(y);
}

// ---------------------------------------------------------------------------
// LayerNorm: one wave per token (float4 per lane). outB (bf16) or outF (f32).
// ---------------------------------------------------------------------------
__global__ void ln_kernel(const float* __restrict__ h,
                          const float* __restrict__ g,
                          const float* __restrict__ b,
                          unsigned short* __restrict__ outB,
                          float* __restrict__ outF) {
    int wave = threadIdx.x >> 5, lane = threadIdx.x & 31;
    int t = blockIdx.x * 8 + wave;             // blockDim = 256 -> 8 waves
    if (t >= TOKENS) return;
    const float4* hp = (const float4*)(h + (size_t)t * D_MODEL);
    float4 v = hp[lane];
    float mean = wave_sum(v.x + v.y + v.z + v.w) * (1.0f / 128.0f);
    float dx = v.x - mean, dy = v.y - mean, dz = v.z - mean, dw = v.w - mean;
    float var = wave_sum(dx * dx + dy * dy + dz * dz + dw * dw) * (1.0f / 128.0f);
    float rs = rsqrtf(var + 1e-5f);
    float4 gv = ((const float4*)g)[lane];
    float4 bv = ((const float4*)b)[lane];
    float ox = dx * rs * gv.x + bv.x;
    float oy = dy * rs * gv.y + bv.y;
    float oz = dz * rs * gv.z + bv.z;
    float ow = dw * rs * gv.w + bv.w;
    if (outB) {
        ushort4 p;
        p.x = f2bf(ox); p.y = f2bf(oy); p.z = f2bf(oz); p.w = f2bf(ow);
        ((ushort4*)(outB + (size_t)t * D_MODEL))[lane] = p;
    } else {
        float4 o; o.x = ox; o.y = oy; o.z = oz; o.w = ow;
        ((float4*)(outF + (size_t)t * D_MODEL))[lane] = o;
    }
}

// ---------------------------------------------------------------------------
// WMMA GEMM, register-blocked 32x64 macro-tile per wave (2 M-tiles x 4 N-tiles).
// A  : (T x K)  bf16 row-major activations
// Wpk: pre-swizzled bf16 B fragments (see pack_w_kernel)
// mode 0: outF = acc                       (GAT projection)
// mode 1: outB = bf16(gelu(acc + bias))    (FFN up)
// mode 2: outF += acc + bias               (FFN down, residual)
// blockDim = 128 (4 waves); grid = (Mpairs/4, N/64)
// Per K-step: 12 b128 loads feed 8 v_wmma (A reused over 4 N-tiles,
// B reused over 2 M-tiles).
// ---------------------------------------------------------------------------
__global__ void gemm_wmma_kernel(const unsigned short* __restrict__ A,
                                 const unsigned short* __restrict__ Wpk,
                                 const float* __restrict__ bias,
                                 float* __restrict__ outF,
                                 unsigned short* __restrict__ outB,
                                 int K, int N, int mode) {
    int wave = threadIdx.x >> 5;
    int lane = threadIdx.x & 31;
    int mt0  = (blockIdx.x * 4 + wave) * 2;     // first of 2 M-tiles
    int nq   = blockIdx.y;                      // quad of 4 N-tiles
    int Kt   = K >> 5;

    int half = lane >> 4;
    int arow = mt0 * 16 + (lane & 15);
    const unsigned short* Ap0 = A + (size_t)arow * K + half * 8;
    const unsigned short* Ap1 = Ap0 + (size_t)16 * K;
    // B fragment base for ntile = nq*4 + ni lives at offset (ntile*Kt*32+lane)*16
    const unsigned short* Bp = Wpk + ((size_t)(nq * 4) * Kt * 32 + lane) * 16;
    const size_t BSTRIDE_N = (size_t)Kt * 512;  // elements per N-tile of frags

    v8f acc[2][4];
    #pragma unroll
    for (int mi = 0; mi < 2; ++mi)
        #pragma unroll
        for (int ni = 0; ni < 4; ++ni)
            acc[mi][ni] = (v8f){};

    for (int kt = 0; kt < Kt; ++kt) {
        Frag16 a0, a1;
        a0.u[0] = *(const u32x4*)(Ap0 + kt * 32);
        a0.u[1] = *(const u32x4*)(Ap0 + kt * 32 + 16);
        a1.u[0] = *(const u32x4*)(Ap1 + kt * 32);
        a1.u[1] = *(const u32x4*)(Ap1 + kt * 32 + 16);
        #pragma unroll
        for (int ni = 0; ni < 4; ++ni) {
            Frag16 bf;
            const unsigned short* bp = Bp + ni * BSTRIDE_N + (size_t)kt * 512;
            bf.u[0] = *(const u32x4*)(bp);
            bf.u[1] = *(const u32x4*)(bp + 8);
            acc[0][ni] = __builtin_amdgcn_wmma_f32_16x16x32_bf16(
                             false, a0.v, false, bf.v, (short)0, acc[0][ni], false, false);
            acc[1][ni] = __builtin_amdgcn_wmma_f32_16x16x32_bf16(
                             false, a1.v, false, bf.v, (short)0, acc[1][ni], false, false);
        }
    }

    // Epilogue: C layout -> lane col = lane&15, rows = half*8 + v
    #pragma unroll
    for (int mi = 0; mi < 2; ++mi) {
        int rbase = (mt0 + mi) * 16 + half * 8;
        #pragma unroll
        for (int ni = 0; ni < 4; ++ni) {
            int ncol = (nq * 4 + ni) * 16 + (lane & 15);
            float bv = bias ? bias[ncol] : 0.0f;
            #pragma unroll
            for (int v = 0; v < 8; ++v) {
                size_t idx = (size_t)(rbase + v) * N + ncol;
                float y = acc[mi][ni][v];
                if (mode == 0) {
                    outF[idx] = y;
                } else if (mode == 1) {
                    outB[idx] = f2bf(gelu_f(y + bv));
                } else {
                    outF[idx] += y + bv;
                }
            }
        }
    }
}

// ---------------------------------------------------------------------------
// GAT edge softmax + aggregation. One wave per (token, head); lane = dim.
// Hex 11x11 neighbors computed analytically (edge list is symmetric).
// h[t, head*32+lane] += sum_k alpha_k * hW[nb_k, head*32+lane]
// ---------------------------------------------------------------------------
__global__ void gat_kernel(const float* __restrict__ hW,
                           const float* __restrict__ a_src,
                           const float* __restrict__ a_dst,
                           float* __restrict__ h) {
    int t    = blockIdx.x;
    int head = threadIdx.x >> 5;
    int lane = threadIdx.x & 31;
    int bb = t / N_CELLS, n = t % N_CELLS;
    int q = n / 11, r = n % 11;

    float ad = a_dst[head * 32 + lane];
    float as = a_src[head * 32 + lane];
    const float* base = hW + (size_t)bb * N_CELLS * D_MODEL + head * 32 + lane;

    float sdst = wave_sum(base[(size_t)n * D_MODEL] * ad);

    const int dq[6] = { 1, -1, 0,  0,  1, -1 };
    const int dr[6] = { 0,  0, 1, -1, -1,  1 };
    float vals[6], e[6];
    int cnt = 0;
    #pragma unroll
    for (int k = 0; k < 6; ++k) {
        int qn = q + dq[k], rn = r + dr[k];
        if ((unsigned)qn < 11u && (unsigned)rn < 11u) {
            int nn = qn * 11 + rn;
            float v = base[(size_t)nn * D_MODEL];
            float s = wave_sum(v * as) + sdst;
            vals[cnt] = v;
            e[cnt] = (s > 0.0f) ? s : 0.2f * s;   // leaky relu 0.2
            ++cnt;
        }
    }
    float m = -1e30f;
    for (int k = 0; k < cnt; ++k) m = fmaxf(m, e[k]);
    float w[6], denom = 0.0f;
    for (int k = 0; k < cnt; ++k) { w[k] = expf(e[k] - m); denom += w[k]; }
    float inv = 1.0f / (denom + 1e-10f);
    float out = 0.0f;
    for (int k = 0; k < cnt; ++k) out += w[k] * inv * vals[k];

    h[(size_t)t * D_MODEL + head * 32 + lane] += out;
}

// ---------------------------------------------------------------------------
// Launcher
// ---------------------------------------------------------------------------
extern "C" void kernel_launch(void* const* d_in, const int* in_sizes, int n_in,
                              void* d_out, int out_size, void* d_ws, size_t ws_size,
                              hipStream_t stream) {
    const float* x       = (const float*)d_in[0];
    const float* enc_W   = (const float*)d_in[1];
    const float* enc_b   = (const float*)d_in[2];
    const float* enc_lng = (const float*)d_in[3];
    const float* enc_lnb = (const float*)d_in[4];
    const float* ln1_g   = (const float*)d_in[5];
    const float* ln1_b   = (const float*)d_in[6];
    const float* gat_W   = (const float*)d_in[7];
    const float* a_src   = (const float*)d_in[8];
    const float* a_dst   = (const float*)d_in[9];
    const float* ln2_g   = (const float*)d_in[10];
    const float* ln2_b   = (const float*)d_in[11];
    const float* ff_W1   = (const float*)d_in[12];
    const float* ff_b1   = (const float*)d_in[13];
    const float* ff_W2   = (const float*)d_in[14];
    const float* ff_b2   = (const float*)d_in[15];
    const float* fn_g    = (const float*)d_in[16];
    const float* fn_b    = (const float*)d_in[17];
    (void)in_sizes; (void)n_in; (void)out_size; (void)ws_size;

    // Workspace layout (256B-aligned chunks). hW (f32) and G (bf16) are
    // time-disjoint, so they share one region.
    char* ws = (char*)d_ws;
    const size_t SZ_H   = (size_t)TOKENS * D_MODEL * sizeof(float);         // 15,859,712
    const size_t SZ_SCR = (size_t)TOKENS * FF_DIM * sizeof(unsigned short); // 31,719,424
    const size_t SZ_HNB = (size_t)TOKENS * D_MODEL * sizeof(unsigned short);//  7,929,856
    float*          h   = (float*)ws;
    float*          hW  = (float*)(ws + SZ_H);                 // aliases G
    unsigned short* G   = (unsigned short*)(ws + SZ_H);
    unsigned short* hnB = (unsigned short*)(ws + SZ_H + SZ_SCR);
    unsigned short* WpG = (unsigned short*)(ws + SZ_H + SZ_SCR + SZ_HNB);
    unsigned short* Wp1 = WpG + (size_t)N_LAYERS * D_MODEL * D_MODEL;
    unsigned short* Wp2 = Wp1 + (size_t)N_LAYERS * D_MODEL * FF_DIM;

    // --- pack all weights to bf16 WMMA fragment layout -------------------
    {
        int tot_g = N_LAYERS * D_MODEL * D_MODEL;   // 98,304
        int tot_f = N_LAYERS * D_MODEL * FF_DIM;    // 393,216
        pack_w_kernel<<<(tot_g + 255) / 256, 256, 0, stream>>>(gat_W, WpG, N_LAYERS, D_MODEL, D_MODEL);
        pack_w_kernel<<<(tot_f + 255) / 256, 256, 0, stream>>>(ff_W1, Wp1, N_LAYERS, D_MODEL, FF_DIM);
        pack_w_kernel<<<(tot_f + 255) / 256, 256, 0, stream>>>(ff_W2, Wp2, N_LAYERS, FF_DIM, D_MODEL);
    }

    // --- encoder ---------------------------------------------------------
    encoder_kernel<<<TOKENS, D_MODEL, 0, stream>>>(x, enc_W, enc_b, enc_lng, enc_lnb, h);

    const int MPAIR4 = (TOKENS / 32) / 4;       // 968 M-pairs -> 242 blocks of 4 waves
    const dim3 gemm_b(128);
    const dim3 ln_grid(TOKENS / 8);

    for (int l = 0; l < N_LAYERS; ++l) {
        // LN1 -> bf16
        ln_kernel<<<ln_grid, 256, 0, stream>>>(h, ln1_g + l * D_MODEL, ln1_b + l * D_MODEL,
                                               hnB, nullptr);
        // hW = hn @ gat_W[l]   (K=128, N=128)
        gemm_wmma_kernel<<<dim3(MPAIR4, D_MODEL / 64), gemm_b, 0, stream>>>(
            hnB, WpG + (size_t)l * D_MODEL * D_MODEL, nullptr, hW, nullptr,
            D_MODEL, D_MODEL, 0);
        // GAT attention + residual into h
        gat_kernel<<<TOKENS, 128, 0, stream>>>(hW, a_src + l * D_MODEL,
                                               a_dst + l * D_MODEL, h);
        // LN2 -> bf16
        ln_kernel<<<ln_grid, 256, 0, stream>>>(h, ln2_g + l * D_MODEL, ln2_b + l * D_MODEL,
                                               hnB, nullptr);
        // G = bf16(gelu(hn @ W1 + b1))   (K=128, N=512)
        gemm_wmma_kernel<<<dim3(MPAIR4, FF_DIM / 64), gemm_b, 0, stream>>>(
            hnB, Wp1 + (size_t)l * D_MODEL * FF_DIM, ff_b1 + l * FF_DIM,
            nullptr, G, D_MODEL, FF_DIM, 1);
        // h += G @ W2 + b2               (K=512, N=128)
        gemm_wmma_kernel<<<dim3(MPAIR4, D_MODEL / 64), gemm_b, 0, stream>>>(
            G, Wp2 + (size_t)l * FF_DIM * D_MODEL, ff_b2 + l * D_MODEL,
            h, nullptr, FF_DIM, D_MODEL, 2);
    }

    // --- final layernorm -> f32 output ----------------------------------
    ln_kernel<<<ln_grid, 256, 0, stream>>>(h, fn_g, fn_b, nullptr, (float*)d_out);
}
